// MemAGDNConv_14173392077041
// MI455X (gfx1250) — matrix operand order repaired and checked by reading, not compiled, not run on hardware
//
#include <hip/hip_runtime.h>

// ---------------- problem constants ----------------
constexpr int   Nn    = 50000;
constexpr int   Ee    = 800000;
constexpr int   Hh    = 2;
constexpr int   DINc  = 256;
constexpr int   Dc    = 64;        // D_
constexpr int   DOUTc = 64;
constexpr int   ROWS  = Nn * Hh;   // 100000 gemm rows
constexpr int   RT    = ROWS / 16; // 6250 row tiles (exact)
#define SLOPE 0.2f

typedef __attribute__((ext_vector_type(16))) __bf16 v16bf;
typedef __attribute__((ext_vector_type(8)))  float  v8f;

// ---------------- helpers ----------------
__device__ __forceinline__ float warp_sum32(float v) {
  v += __shfl_xor(v, 16, 32);
  v += __shfl_xor(v, 8, 32);
  v += __shfl_xor(v, 4, 32);
  v += __shfl_xor(v, 2, 32);
  v += __shfl_xor(v, 1, 32);
  return v;
}
// monotonic float<->int mapping so segment-max uses native global_atomic_max_i32
__device__ __forceinline__ int ford(float f) {
  int i = __float_as_int(f);
  return i >= 0 ? i : (i ^ 0x7fffffff);
}
__device__ __forceinline__ float funord(int i) {
  return __int_as_float(i >= 0 ? i : (i ^ 0x7fffffff));
}
// round-to-nearest-even f32 -> bf16
__device__ __forceinline__ unsigned short f2bf(float f) {
  unsigned int u = __float_as_uint(f);
  u += 0x7fffu + ((u >> 16) & 1u);
  return (unsigned short)(u >> 16);
}
// native f32 atomic add (global_atomic_add_f32, no CAS loop)
__device__ __forceinline__ void atomic_add_f32(float* p, float v) {
  __hip_atomic_fetch_add(p, v, __ATOMIC_RELAXED, __HIP_MEMORY_SCOPE_AGENT);
}

// ---------------- 1: per-node el/er dots + x0 init ----------------
// one wave (32 lanes) per node; lane l holds feat[n, 8l .. 8l+7]
__global__ void k_node_pre(const float* __restrict__ feat,
                           const float* __restrict__ attn_l,
                           const float* __restrict__ attn_r,
                           const float* __restrict__ pe,
                           float* __restrict__ el, float* __restrict__ er,
                           float* __restrict__ x0) {
  const int lane = threadIdx.x & 31;
  const int n = blockIdx.x * 8 + (threadIdx.x >> 5);
  if (n >= Nn) return;

  const float4* fp = (const float4*)(feat + (size_t)n * DINc);
  float4 f0 = fp[lane * 2];
  float4 f1 = fp[lane * 2 + 1];

  const float4* al0 = (const float4*)(attn_l);
  const float4* al1 = (const float4*)(attn_l + DINc);
  const float4* ar0 = (const float4*)(attn_r);
  const float4* ar1 = (const float4*)(attn_r + DINc);

  float dl0, dl1, dr0, dr1;
  {
    float4 a = al0[lane * 2], b = al0[lane * 2 + 1];
    dl0 = f0.x*a.x + f0.y*a.y + f0.z*a.z + f0.w*a.w
        + f1.x*b.x + f1.y*b.y + f1.z*b.z + f1.w*b.w;
    a = al1[lane * 2]; b = al1[lane * 2 + 1];
    dl1 = f0.x*a.x + f0.y*a.y + f0.z*a.z + f0.w*a.w
        + f1.x*b.x + f1.y*b.y + f1.z*b.z + f1.w*b.w;
    a = ar0[lane * 2]; b = ar0[lane * 2 + 1];
    dr0 = f0.x*a.x + f0.y*a.y + f0.z*a.z + f0.w*a.w
        + f1.x*b.x + f1.y*b.y + f1.z*b.z + f1.w*b.w;
    a = ar1[lane * 2]; b = ar1[lane * 2 + 1];
    dr1 = f0.x*a.x + f0.y*a.y + f0.z*a.z + f0.w*a.w
        + f1.x*b.x + f1.y*b.y + f1.z*b.z + f1.w*b.w;
  }
  dl0 = warp_sum32(dl0); dl1 = warp_sum32(dl1);
  dr0 = warp_sum32(dr0); dr1 = warp_sum32(dr1);

  // xsum[d] = sum_k feat[n, k*64+d]; lanes differing in bits 3,4 hold same d-range
  float s[8] = {f0.x, f0.y, f0.z, f0.w, f1.x, f1.y, f1.z, f1.w};
  #pragma unroll
  for (int j = 0; j < 8; ++j) {
    s[j] += __shfl_xor(s[j], 8, 32);
    s[j] += __shfl_xor(s[j], 16, 32);
  }
  if (lane < 16) {
    const int h = lane >> 3;               // lanes 0-7 -> h=0, 8-15 -> h=1
    const int dbase = (lane & 7) * 8;
    float* xp = x0 + (size_t)n * (Hh * Dc) + h * Dc + dbase;
    const float* pp = pe + h * ((4 + 1) * Dc) + dbase;   // pe[h][0][d]
    #pragma unroll
    for (int j = 0; j < 8; ++j) xp[j] = s[j] + pp[j];
  }
  if (lane == 0) {
    el[n * 2 + 0] = dl0; el[n * 2 + 1] = dl1;
    er[n * 2 + 0] = dr0; er[n * 2 + 1] = dr1;
  }
}

// ---------------- 2: init per-(node,head) softmax state ----------------
__global__ void k_init_nh(int* __restrict__ emax_i, float* __restrict__ denom) {
  int i = blockIdx.x * blockDim.x + threadIdx.x;
  if (i < ROWS) { emax_i[i] = (int)0x80000000; denom[i] = 0.f; }
}

// ---------------- 3: edge segment-max ----------------
__global__ void k_edge_max(const int* __restrict__ src, const int* __restrict__ dst,
                           const float* __restrict__ el, const float* __restrict__ er,
                           int* __restrict__ emax_i) {
  int e = blockIdx.x * blockDim.x + threadIdx.x;
  if (e >= Ee) return;
  int s = src[e], d = dst[e];
  #pragma unroll
  for (int h = 0; h < 2; ++h) {
    float v = el[s * 2 + h] + er[d * 2 + h];
    v = v > 0.f ? v : SLOPE * v;
    atomicMax(&emax_i[d * 2 + h], ford(v));
  }
}

// ---------------- 4: ee = exp(e - emax), denom = segment_sum ----------------
__global__ void k_edge_ee(const int* __restrict__ src, const int* __restrict__ dst,
                          const float* __restrict__ el, const float* __restrict__ er,
                          const int* __restrict__ emax_i,
                          float* __restrict__ denom, float* __restrict__ ee) {
  int e = blockIdx.x * blockDim.x + threadIdx.x;
  if (e >= Ee) return;
  int s = src[e], d = dst[e];
  #pragma unroll
  for (int h = 0; h < 2; ++h) {
    float v = el[s * 2 + h] + er[d * 2 + h];
    v = v > 0.f ? v : SLOPE * v;
    float x = __expf(v - funord(emax_i[d * 2 + h]));
    ee[e * 2 + h] = x;
    atomic_add_f32(&denom[d * 2 + h], x);
  }
}

// ---------------- 5: a = ee / denom[dst] ----------------
__global__ void k_edge_div(const int* __restrict__ dst,
                           const float* __restrict__ denom, float* __restrict__ a) {
  int i = blockIdx.x * blockDim.x + threadIdx.x;
  if (i >= Ee * Hh) return;
  int e = i >> 1, h = i & 1;
  a[i] = a[i] / denom[dst[e] * 2 + h];
}

// ---------------- 6: zero agg ----------------
__global__ void k_zero(float* __restrict__ p, int n) {
  int i = blockIdx.x * blockDim.x + threadIdx.x;
  if (i < n) p[i] = 0.f;
}

// ---------------- 7: scatter msg = a * x[src] into agg (one wave / edge) ----------------
__global__ void k_scatter(const int* __restrict__ src, const int* __restrict__ dst,
                          const float* __restrict__ a, const float* __restrict__ x,
                          float* __restrict__ agg) {
  const int lane = threadIdx.x & 31;
  const int e = blockIdx.x * 8 + (threadIdx.x >> 5);
  if (e >= Ee) return;
  const int s = src[e], d = dst[e];
  const int h = lane >> 4, q = lane & 15;
  const float av = a[e * 2 + h];
  const float4 xv = *(const float4*)(x + (size_t)s * 128 + h * 64 + q * 4);
  float* ap = agg + (size_t)d * 128 + h * 64 + q * 4;
  atomic_add_f32(ap + 0, av * xv.x);
  atomic_add_f32(ap + 1, av * xv.y);
  atomic_add_f32(ap + 2, av * xv.z);
  atomic_add_f32(ap + 3, av * xv.w);
}

// ---------------- 8: x = agg + xs[:,hop] + pe[:,hop+1]; also emit bf16 hcat ----------------
__global__ void k_update(const float* __restrict__ agg, const float* __restrict__ feat,
                         const float* __restrict__ pe, int hop,
                         float* __restrict__ x, unsigned short* __restrict__ hcat) {
  int i = blockIdx.x * blockDim.x + threadIdx.x;   // over N*H*64
  if (i >= Nn * 128) return;
  int n = i >> 7, r = i & 127, h = r >> 6, dd = r & 63;
  float v = agg[i] + feat[(size_t)n * DINc + hop * 64 + dd]
                   + pe[h * 320 + (hop + 1) * 64 + dd];
  x[i] = v;
  // hcat row = n*2+h, col = hop*64+dd  (row-major stride 256, bf16)
  hcat[(size_t)n * 512 + h * 256 + hop * 64 + dd] = f2bf(v);
}

// ---------------- 9: fc_w f32 -> bf16 ----------------
__global__ void k_wconv(const float* __restrict__ w, unsigned short* __restrict__ wbf) {
  int i = blockIdx.x * blockDim.x + threadIdx.x;
  if (i < DOUTc * DINc) wbf[i] = f2bf(w[i]);
}

// ---------------- 10: WMMA GEMM: out(100000x64) = hcat(100000x256,bf16) * Wt ----------------
// one wave per 16-row tile; 4 column tiles of 16; K=256 in steps of 32
__global__ void k_gemm(const unsigned short* __restrict__ hcat,
                       const unsigned short* __restrict__ wbf,
                       const float* __restrict__ bias,
                       float* __restrict__ out) {
  const int lane = threadIdx.x & 31;
  const int tile = blockIdx.x * 8 + (threadIdx.x >> 5);
  if (tile >= RT) return;                      // wave-uniform guard (EXEC all-1 inside)
  const int rowBase = tile * 16;
  const int hi = lane >> 4;                    // half-wave select
  const int r  = lane & 15;

  v8f acc[4] = {};

  // A layout (16-bit 16x32): lane<16 holds K {0..7,16..23}, lane>=16 holds K {8..15,24..31}
  const unsigned short* arow = hcat + (size_t)(rowBase + r) * 256;

  #pragma unroll
  for (int k0 = 0; k0 < 256; k0 += 32) {
    union { v16bf v; uint4 u[2]; } A;
    A.u[0] = *(const uint4*)(arow + k0 + hi * 8);
    A.u[1] = *(const uint4*)(arow + k0 + 16 + hi * 8);
    #pragma unroll
    for (int c = 0; c < 4; ++c) {
      // B layout (16-bit 32x16): lane<16 = col r, K {0..7,16..23}; lane>=16 = col r, K {8..15,24..31}
      // B[k][n] = fc_w[n][k] -> halves k0+8*hi+{0..7,16..23} from fc_w row (c*16+r)
      union { v16bf v; uint4 u[2]; } B;
      const unsigned short* brow = wbf + (size_t)(c * 16 + r) * 256 + k0 + hi * 8;
      B.u[0] = ((const uint4*)brow)[0];
      B.u[1] = ((const uint4*)brow)[2];
      acc[c] = __builtin_amdgcn_wmma_f32_16x16x32_bf16(
          false, A.v, false, B.v, (short)0, acc[c], false, false);
    }
  }

  // C/D layout: VGPR j -> row = rowBase + hi*8 + j, col = c*16 + r
  #pragma unroll
  for (int c = 0; c < 4; ++c) {
    const int col = c * 16 + r;
    const float b0 = bias[col];        // h = row&1 = j&1 (rowBase, hi*8 even)
    const float b1 = bias[64 + col];
    #pragma unroll
    for (int j = 0; j < 8; ++j) {
      const int row = rowBase + hi * 8 + j;
      out[(size_t)row * 64 + col] = acc[c][j] + ((j & 1) ? b1 : b0);
    }
  }
}

// ---------------- host ----------------
extern "C" void kernel_launch(void* const* d_in, const int* in_sizes, int n_in,
                              void* d_out, int out_size, void* d_ws, size_t ws_size,
                              hipStream_t stream) {
  const float* feat   = (const float*)d_in[0];
  const int*   src    = (const int*)  d_in[1];
  const int*   dst    = (const int*)  d_in[2];
  const float* fc_w   = (const float*)d_in[3];
  const float* attn_l = (const float*)d_in[4];
  const float* attn_r = (const float*)d_in[5];
  const float* pe     = (const float*)d_in[6];
  const float* bias   = (const float*)d_in[7];
  float* out = (float*)d_out;

  char* ws = (char*)d_ws;
  size_t off = 0;
  auto take = [&](size_t bytes) -> char* {
    char* p = ws + off;
    off = (off + bytes + 255) & ~(size_t)255;
    return p;
  };
  float*          el     = (float*)take((size_t)ROWS * 4);
  float*          er     = (float*)take((size_t)ROWS * 4);
  int*            emax_i = (int*)  take((size_t)ROWS * 4);
  float*          denom  = (float*)take((size_t)ROWS * 4);
  float*          a      = (float*)take((size_t)Ee * Hh * 4);          // 6.4 MB (ee then a)
  float*          x      = (float*)take((size_t)Nn * 128 * 4);         // 25.6 MB
  float*          agg    = (float*)take((size_t)Nn * 128 * 4);         // 25.6 MB
  unsigned short* hcat   = (unsigned short*)take((size_t)ROWS * 256 * 2); // 51.2 MB
  unsigned short* wbf    = (unsigned short*)take((size_t)DOUTc * DINc * 2);
  (void)in_sizes; (void)n_in; (void)out_size; (void)ws_size;

  k_node_pre<<<Nn / 8, 256, 0, stream>>>(feat, attn_l, attn_r, pe, el, er, x);
  k_init_nh<<<(ROWS + 255) / 256, 256, 0, stream>>>(emax_i, denom);
  k_wconv<<<(DOUTc * DINc + 255) / 256, 256, 0, stream>>>(fc_w, wbf);

  k_edge_max<<<(Ee + 255) / 256, 256, 0, stream>>>(src, dst, el, er, emax_i);
  k_edge_ee <<<(Ee + 255) / 256, 256, 0, stream>>>(src, dst, el, er, emax_i, denom, a);
  k_edge_div<<<(Ee * Hh + 255) / 256, 256, 0, stream>>>(dst, denom, a);

  for (int hop = 0; hop < 4; ++hop) {
    k_zero   <<<(Nn * 128 + 255) / 256, 256, 0, stream>>>(agg, Nn * 128);
    k_scatter<<<Ee / 8, 256, 0, stream>>>(src, dst, a, x, agg);
    k_update <<<(Nn * 128 + 255) / 256, 256, 0, stream>>>(agg, feat, pe, hop, x, hcat);
  }

  k_gemm<<<(RT + 7) / 8, 256, 0, stream>>>(hcat, wbf, bias, out);
}
